// EdgeDecoder_61503931678932
// MI455X (gfx1250) — matrix operand order repaired and checked by reading, not compiled, not run on hardware
//
#include <hip/hip_runtime.h>
#include <hip/hip_bf16.h>
#include <math.h>

// MI455X / gfx1250: wave32, WMMA 16x16x32 f16->f32.
// EdgeDecoder: per-edge MLP (259->128 relu, LN(128), 128->1).
// GEMM on WMMA; W1 pre-transposed to f16 in workspace by a prepass kernel;
// features gathered+f16-converted through LDS; LN epilogue fully in registers.

typedef __attribute__((ext_vector_type(16))) _Float16 v16h;
typedef __attribute__((ext_vector_type(8)))  float    v8f;

#define H_DIM   128
#define IN_DIM  259
#define KP      296              // padded K stride in halves (288 rounded up, 16B-aligned rows)
#define KSTEPS  9                // ceil(259/32) -> 288
#define NTILES  8                // 128 / 16 output column tiles
#define WAVES   4
#define TPW     4                // 16-edge tiles per wave
#define THREADS (WAVES * 32)
#define TILES_PER_BLOCK (WAVES * TPW)
#define W1T_HALVES (H_DIM * KP)  // 37,888 halves = 75,776 bytes

union AF { v16h h; uint4 u[2]; };

// ---- Prepass: W1 [259,128] f32  ->  W1T [128][296] f16 (zero-padded K) in d_ws ----
__global__ void EdgeDecoder_w1t_prep(const float* __restrict__ W1,
                                     _Float16* __restrict__ W1T) {
  const int idx = blockIdx.x * 256 + threadIdx.x;    // linear over k-major [KP x H]
  if (idx >= KP * H_DIM) return;
  const int k = idx / H_DIM;
  const int n = idx - k * H_DIM;
  _Float16 v = (_Float16)0.f;
  if (k < IN_DIM) v = (_Float16)W1[idx];             // W1[k*H + n] == W1[idx], coalesced
  W1T[n * KP + k] = v;
}

__global__ __launch_bounds__(THREADS, 1)
void EdgeDecoder_61503931678932_kernel(
    const float* __restrict__ hf,            // [N,128]
    const long long* __restrict__ pe,        // [2,E] int64
    const float* __restrict__ onsets,
    const float* __restrict__ durations,
    const float* __restrict__ pitches,
    const float* __restrict__ onset_beat,
    const float* __restrict__ duration_beat,
    const float* __restrict__ ts_beats,
    const float* __restrict__ W1,            // [259,128]
    const float* __restrict__ b1,            // [128]
    const float* __restrict__ gmm,           // [128]
    const float* __restrict__ bta,           // [128]
    const float* __restrict__ W2,            // [128,1]
    const float* __restrict__ b2,            // [1]
    const _Float16* __restrict__ w1t,        // pre-transposed f16 W1 (or null)
    float* __restrict__ out,                 // [E]
    int E, int NT)
{
  __shared__ _Float16 sW[W1T_HALVES];        // W1 transposed [n][k] in f16 (75,776 B)
  __shared__ _Float16 sA[WAVES * 16 * KP];   // per-wave A tiles  (37,888 B)

  const int tid  = threadIdx.x;
  const int lane = tid & 31;
  const int wave = tid >> 5;
  const int lm   = lane & 15;
  const int half = lane >> 4;

  // ---- Stage W1T into LDS ----
  if (w1t) {
    const uint4* __restrict__ src = (const uint4*)w1t;
    uint4* dst = (uint4*)sW;
#pragma unroll 4
    for (int i = tid; i < W1T_HALVES / 8; i += THREADS) dst[i] = src[i];
  } else {
    // fallback: convert in-kernel (workspace too small)
    const int n = tid;                       // THREADS == H_DIM
    for (int k = 0; k < KP; ++k) {
      _Float16 v = (_Float16)0.f;
      if (k < IN_DIM) v = (_Float16)W1[(long long)k * H_DIM + n];
      sW[n * KP + k] = v;
    }
  }

  // ---- Per-lane epilogue constants: n = t*16 + lm matches C-fragment lanes ----
  float b1v[NTILES], gv[NTILES], bev[NTILES], w2v[NTILES];
#pragma unroll
  for (int t = 0; t < NTILES; ++t) {
    const int n = t * 16 + lm;
    b1v[t] = b1[n]; gv[t] = gmm[n]; bev[t] = bta[n]; w2v[t] = W2[n];
  }
  const float b2v = b2[0];

  __syncthreads();

  _Float16* myA = sA + wave * 16 * KP;

  for (int tt = 0; tt < TPW; ++tt) {
    const int g = blockIdx.x * TILES_PER_BLOCK + wave * TPW + tt;   // 16-edge tile id
    const bool valid = (g < NT);
    const long long e0 = (long long)g * 16;

    // ---- Stage A tile: 16 edges x 288 features, f16, via coalesced float4 gathers ----
    if (valid) {
      int rIdx = 0, cIdx = 0;
      if (lane < 16) {
        long long e = e0 + lane; if (e >= E) e = E - 1;
        rIdx = (int)pe[e];
        cIdx = (int)pe[(long long)E + e];
      }
      for (int e = 0; e < 16; ++e) {
        const int r = __shfl(rIdx, e);
        const int c = __shfl(cIdx, e);
#pragma unroll
        for (int j = 0; j < 3; ++j) {
          const int k = (lane + 32 * j) * 4;
          if (k < KP) {
            float4 f;
            if (k < H_DIM) {
              f = *(const float4*)(hf + (long long)r * H_DIM + k);
            } else if (k < 2 * H_DIM) {
              f = *(const float4*)(hf + (long long)c * H_DIM + (k - H_DIM));
            } else if (k == 2 * H_DIM) {
              const float offb = onset_beat[r] + duration_beat[r];
              const float cont = 1.f - tanhf((onset_beat[c] - offb) / ts_beats[c]);
              const float offs = onsets[r] + durations[r];
              const float bin  = (onsets[c] == offs) ? 1.f : 0.f;
              const float ps   = fabsf(pitches[c] - pitches[r]) * (1.f / 127.f);
              f = make_float4(cont, bin, ps, 0.f);
            } else {
              f = make_float4(0.f, 0.f, 0.f, 0.f);   // zero pad (avoid NaN*0 in WMMA)
            }
            union { _Float16 h[4]; uint2 u; } pk;
            pk.h[0] = (_Float16)f.x; pk.h[1] = (_Float16)f.y;
            pk.h[2] = (_Float16)f.z; pk.h[3] = (_Float16)f.w;
            *(uint2*)(myA + e * KP + k) = pk.u;
          }
        }
      }
    }

    __syncthreads();

    if (valid) {
      // ---- GEMM: 16(edges) x 128(hidden), K = 288 in 9 steps of 32 ----
      v8f acc[NTILES];
#pragma unroll
      for (int t = 0; t < NTILES; ++t)
#pragma unroll
        for (int j = 0; j < 8; ++j) acc[t][j] = 0.f;

      const _Float16* aRow = myA + lm * KP;  // lane's edge row
      auto loadA = [&](int kk) {
        AF x;
        const _Float16* p = aRow + kk * 32 + 8 * half;
        x.u[0] = *(const uint4*)p;           // K 0..7  / 8..15
        x.u[1] = *(const uint4*)(p + 16);    // K 16..23/ 24..31
        return x;
      };
      auto loadB = [&](int kk, int t) {
        AF x;
        const _Float16* p = sW + (t * 16 + lm) * KP + kk * 32 + 8 * half;
        x.u[0] = *(const uint4*)p;
        x.u[1] = *(const uint4*)(p + 16);
        return x;
      };

      // one-stage software pipeline: operands for WMMA[t] fetched one step ahead
      AF a = loadA(0);
      AF b = loadB(0, 0);
#pragma unroll
      for (int kk = 0; kk < KSTEPS; ++kk) {
        AF an = a;
        if (kk + 1 < KSTEPS) an = loadA(kk + 1);
#pragma unroll
        for (int t = 0; t < NTILES; ++t) {
          AF bn = b;
          if (t + 1 < NTILES)           bn = loadB(kk, t + 1);
          else if (kk + 1 < KSTEPS)     bn = loadB(kk + 1, 0);
          acc[t] = __builtin_amdgcn_wmma_f32_16x16x32_f16(
              false, a.h, false, b.h, (short)0, acc[t], false, false);
          b = bn;
        }
        a = an;
      }

      // ---- Epilogue in registers: bias + ReLU ----
#pragma unroll
      for (int t = 0; t < NTILES; ++t)
#pragma unroll
        for (int j = 0; j < 8; ++j) {
          const float v = acc[t][j] + b1v[t];
          acc[t][j] = v > 0.f ? v : 0.f;
        }

      // LayerNorm stats: reduce across 8 tiles (regs) then 16 lanes (shfl_xor within half)
      float mean[8], rstd[8];
#pragma unroll
      for (int j = 0; j < 8; ++j) {
        float s = 0.f;
#pragma unroll
        for (int t = 0; t < NTILES; ++t) s += acc[t][j];
        s += __shfl_xor(s, 1); s += __shfl_xor(s, 2);
        s += __shfl_xor(s, 4); s += __shfl_xor(s, 8);
        mean[j] = s * (1.f / 128.f);
      }
#pragma unroll
      for (int j = 0; j < 8; ++j) {
        float v = 0.f;
#pragma unroll
        for (int t = 0; t < NTILES; ++t) { const float d = acc[t][j] - mean[j]; v += d * d; }
        v += __shfl_xor(v, 1); v += __shfl_xor(v, 2);
        v += __shfl_xor(v, 4); v += __shfl_xor(v, 8);
        rstd[j] = rsqrtf(v * (1.f / 128.f) + 1e-5f);
      }

      // LN affine + dot with W2 fused
      float res[8];
#pragma unroll
      for (int j = 0; j < 8; ++j) {
        float o = 0.f;
#pragma unroll
        for (int t = 0; t < NTILES; ++t) {
          const float hn = (acc[t][j] - mean[j]) * rstd[j] * gv[t] + bev[t];
          o += hn * w2v[t];
        }
        o += __shfl_xor(o, 1); o += __shfl_xor(o, 2);
        o += __shfl_xor(o, 4); o += __shfl_xor(o, 8);
        res[j] = o + b2v;
      }

      // VGPR j holds edges e0+j (lanes 0-15) and e0+8+j (lanes 16-31)
      if (lm == 0) {
        const long long base = e0 + (long long)half * 8;
#pragma unroll
        for (int j = 0; j < 8; ++j) {
          const long long e = base + j;
          if (e < E) out[e] = res[j];
        }
      }
    }

    __syncthreads();
  }
}

extern "C" void kernel_launch(void* const* d_in, const int* in_sizes, int n_in,
                              void* d_out, int out_size, void* d_ws, size_t ws_size,
                              hipStream_t stream) {
  const float*     hf            = (const float*)d_in[0];
  const long long* pe            = (const long long*)d_in[1];
  const float*     onsets        = (const float*)d_in[2];
  const float*     durations     = (const float*)d_in[3];
  const float*     pitches       = (const float*)d_in[4];
  const float*     onset_beat    = (const float*)d_in[5];
  const float*     duration_beat = (const float*)d_in[6];
  const float*     ts_beats      = (const float*)d_in[7];
  const float*     W1            = (const float*)d_in[8];
  const float*     b1            = (const float*)d_in[9];
  const float*     gmm           = (const float*)d_in[10];
  const float*     bta           = (const float*)d_in[11];
  const float*     W2            = (const float*)d_in[12];
  const float*     b2            = (const float*)d_in[13];

  const int E  = in_sizes[1] / 2;
  const int NT = (E + 15) / 16;
  const int blocks = (NT + TILES_PER_BLOCK - 1) / TILES_PER_BLOCK;

  _Float16* w1t = nullptr;
  if (ws_size >= (size_t)W1T_HALVES * sizeof(_Float16)) {
    w1t = (_Float16*)d_ws;
    const int total = KP * H_DIM;
    EdgeDecoder_w1t_prep<<<(total + 255) / 256, 256, 0, stream>>>(W1, w1t);
  }

  EdgeDecoder_61503931678932_kernel<<<blocks, THREADS, 0, stream>>>(
      hf, pe, onsets, durations, pitches, onset_beat, duration_beat, ts_beats,
      W1, b1, gmm, bta, W2, b2, w1t, (float*)d_out, E, NT);
}
